// TextMoE_44719199486753
// MI455X (gfx1250) — matrix operand order repaired and checked by reading, not compiled, not run on hardware
//
#include <hip/hip_runtime.h>
#include <hip/hip_bf16.h>

// ---------------------------------------------------------------------------
// TextMoE forward for MI455X (gfx1250): bf16 WMMA GEMMs + fp32 VALU epilogues
// ---------------------------------------------------------------------------

typedef __attribute__((ext_vector_type(16))) __bf16        v16bf;
typedef __attribute__((ext_vector_type(8)))  float         v8f;
typedef __attribute__((ext_vector_type(8)))  unsigned int  v8u;

#define Bsz   16
#define Ssz   512
#define Dsz   1024
#define Hsz   8
#define HDsz  128
#define Esz   7
#define DFsz  4096
#define ROWS  (Bsz * Ssz)   // 8192

// GEMM block tile: 128(M) x 128(N), K-step 32; 256 threads = 8 wave32 waves,
// each wave owns a 32(M) x 64(N) sub-tile = 2x4 v_wmma_f32_16x16x32_bf16.
#define BM 128
#define BN 128
#define BK 32
#define LDP (BK + 8)        // LDS row pitch in shorts (80 B: 16B-aligned rows)

__device__ __forceinline__ unsigned short f2bf(float f) {
  unsigned int u = __float_as_uint(f);
  u += 0x7FFFu + ((u >> 16) & 1u);       // round-to-nearest-even
  return (unsigned short)(u >> 16);
}
__device__ __forceinline__ unsigned int pack2bf(float a, float b) {
  return (unsigned int)f2bf(a) | ((unsigned int)f2bf(b) << 16);
}

__device__ __forceinline__ float gelu_tanh(float x) {
  const float k0 = 0.7978845608028654f;  // sqrt(2/pi)
  float x3 = x * x * x;
  return 0.5f * x * (1.0f + tanhf(k0 * (x + 0.044715f * x3)));
}

__device__ __forceinline__ float waveSum(float v) {
#pragma unroll
  for (int m = 16; m; m >>= 1) v += __shfl_xor(v, m, 32);
  return v;
}
__device__ __forceinline__ float waveMax(float v) {
#pragma unroll
  for (int m = 16; m; m >>= 1) v = fmaxf(v, __shfl_xor(v, m, 32));
  return v;
}

// Load one 16-element bf16 fragment from LDS (two 16B-aligned b128 reads).
__device__ __forceinline__ v16bf frag16(const unsigned short* p0,
                                        const unsigned short* p1) {
  uint4 lo = *(const uint4*)p0;
  uint4 hi = *(const uint4*)p1;
  v8u u;
  u[0] = lo.x; u[1] = lo.y; u[2] = lo.z; u[3] = lo.w;
  u[4] = hi.x; u[5] = hi.y; u[6] = hi.z; u[7] = hi.w;
  return __builtin_bit_cast(v16bf, u);
}

enum { EPI_STORE = 0, EPI_RESID = 1, EPI_GELU = 2, EPI_MASK_INIT = 3, EPI_MASK_ADD = 4 };

// ---------------------------------------------------------------------------
// Generic batched WMMA GEMM: C = epi(alpha * A x B (+bias)), A fp32 [M,K],
// B fp32 [K,N] (or [N,K] if BT), bf16-converted in LDS, f32 accumulate.
// AVEC: A rows are 16B-aligned (float4 staging); false for the attn@V GEMM
// whose A lives at a 4B-aligned d_out offset.
// ---------------------------------------------------------------------------
template <bool BT, int EPI, bool AVEC>
__global__ __launch_bounds__(256) void gemm_wmma_kernel(
    const float* __restrict__ A, const float* __restrict__ B,
    const float* __restrict__ bias, const float* __restrict__ resid,
    const float* __restrict__ rowScale, int rowScaleStride,
    float* __restrict__ C,
    int Kdim, int lda, int ldb, int ldc, float alpha, int batchH,
    long long sAb, long long sAh, long long sBb, long long sBh,
    long long sCb, long long sCh)
{
  __shared__ unsigned short As[BM][LDP];   // M-major, K contiguous
  __shared__ unsigned short Bs[BN][LDP];   // N-major, K contiguous

  const int z  = blockIdx.z;
  const int zb = z / batchH, zh = z - zb * batchH;
  const float* Ab = A + zb * sAb + zh * sAh;
  const float* Bb = B + zb * sBb + zh * sBh;
  float*       Cb = C + zb * sCb + zh * sCh;

  const int bm   = blockIdx.y * BM;
  const int bn   = blockIdx.x * BN;
  const int tid  = threadIdx.x;
  const int lane = tid & 31;
  const int wid  = tid >> 5;
  const int waveM = wid & 3;   // 4 waves along M (4*32 = 128)
  const int waveN = wid >> 2;  // 2 waves along N (2*64 = 128)

  v8f acc[2][4] = {};

  for (int k0 = 0; k0 < Kdim; k0 += BK) {
    if (k0 + BK < Kdim) {  // prefetch next K tile -> global_prefetch_b8
      __builtin_prefetch(Ab + (long long)(bm + (tid & 127)) * lda + (k0 + BK), 0, 1);
      if (BT)
        __builtin_prefetch(Bb + (long long)(bn + (tid & 127)) * ldb + (k0 + BK), 0, 1);
      else
        __builtin_prefetch(Bb + (long long)(k0 + BK + (tid & 31)) * ldb + bn, 0, 1);
    }

    // ---- stage A (BM x BK), fp32 -> bf16, K-contiguous in LDS ----
    if (AVEC) {
#pragma unroll
      for (int i = 0; i < (BM * BK) / (256 * 4); ++i) {   // 4 float4 per thread
        int e = tid + i * 256;
        int r = e >> 3;                 // 8 float4 per 32-wide row
        int c = (e & 7) << 2;
        float4 f = *(const float4*)(Ab + (long long)(bm + r) * lda + (k0 + c));
        *(uint2*)&As[r][c] = make_uint2(pack2bf(f.x, f.y), pack2bf(f.z, f.w));
      }
    } else {
#pragma unroll
      for (int i = 0; i < (BM * BK) / 256; ++i) {
        int e = tid + i * 256;
        int r = e >> 5, c = e & 31;
        As[r][c] = f2bf(Ab[(long long)(bm + r) * lda + (k0 + c)]);
      }
    }

    // ---- stage B -> Bs[n][k] (transposed to N-major) ----
    if (BT) {
      // B is [N,K] row-major: contiguous along K, vector load + vector store.
#pragma unroll
      for (int i = 0; i < (BN * BK) / (256 * 4); ++i) {   // 4 float4 per thread
        int e = tid + i * 256;
        int r = (e & 7) << 2;           // K offset
        int c = e >> 3;                 // N offset
        float4 f = *(const float4*)(Bb + (long long)(bn + c) * ldb + (k0 + r));
        *(uint2*)&Bs[c][r] = make_uint2(pack2bf(f.x, f.y), pack2bf(f.z, f.w));
      }
    } else {
      // B is [K,N] row-major: coalesced float4 along N, transposed b16 stores.
#pragma unroll
      for (int i = 0; i < (BN * BK) / (256 * 4); ++i) {
        int e = tid + i * 256;
        int c = (e & 31) << 2;          // N offset (BN/4 = 32 float4 per row)
        int r = e >> 5;                 // K offset
        float4 f = *(const float4*)(Bb + (long long)(k0 + r) * ldb + (bn + c));
        Bs[c + 0][r] = f2bf(f.x);
        Bs[c + 1][r] = f2bf(f.y);
        Bs[c + 2][r] = f2bf(f.z);
        Bs[c + 3][r] = f2bf(f.w);
      }
    }
    __syncthreads();

    // ---- fragments per CDNA5 ISA 7.12.2 (16-bit A 16x32 / B 32x16) ----
    const int arow = waveM * 32 + (lane & 15);
    const int kh   = (lane >> 4) << 3;   // lanes 0-15: K0-7/16-23; 16-31: K8-15/24-31
    const int bcol = waveN * 64 + (lane & 15);
    const int kb   = (lane >> 4) << 4;   // lanes 0-15: K0-15; 16-31: K16-31
    v16bf af[2], bf[4];
#pragma unroll
    for (int tm = 0; tm < 2; ++tm) {
      int r = arow + tm * 16;
      af[tm] = frag16(&As[r][kh], &As[r][16 + kh]);
    }
#pragma unroll
    for (int tn = 0; tn < 4; ++tn) {
      int c = bcol + tn * 16;
      bf[tn] = frag16(&Bs[c][kh], &Bs[c][16 + kh]);
    }
#pragma unroll
    for (int tm = 0; tm < 2; ++tm)
#pragma unroll
      for (int tn = 0; tn < 4; ++tn)
        acc[tm][tn] = __builtin_amdgcn_wmma_f32_16x16x32_bf16(
            false, af[tm], false, bf[tn], (short)0, acc[tm][tn], false, false);
    __syncthreads();
  }

  // ---- epilogue; C/D layout: row = 8*(lane>=16) + vgpr, col = lane & 15 ----
#pragma unroll
  for (int tm = 0; tm < 2; ++tm) {
#pragma unroll
    for (int tn = 0; tn < 4; ++tn) {
#pragma unroll
      for (int i = 0; i < 8; ++i) {
        int row = bm + waveM * 32 + tm * 16 + ((lane >> 4) << 3) + i;
        int col = bn + waveN * 64 + tn * 16 + (lane & 15);
        float v = acc[tm][tn][i] * alpha;
        if (bias) v += bias[col];
        if (EPI == EPI_GELU)  v = gelu_tanh(v);
        if (EPI == EPI_RESID) v += resid[(long long)row * ldc + col];
        long long cidx = (long long)row * ldc + col;
        if (EPI == EPI_MASK_INIT) {
          Cb[cidx] = v * rowScale[(long long)row * rowScaleStride];
        } else if (EPI == EPI_MASK_ADD) {
          Cb[cidx] += v * rowScale[(long long)row * rowScaleStride];
        } else {
          Cb[cidx] = v;
        }
      }
    }
  }
}

// ---------------------------------------------------------------------------
// Elementwise / reduction kernels (fp32)
// ---------------------------------------------------------------------------
__global__ __launch_bounds__(256) void embed_kernel(
    const int* __restrict__ ids, const float* __restrict__ emb,
    const float* __restrict__ pos, float* __restrict__ X)
{
  long long t = (long long)blockIdx.x * 256 + threadIdx.x;   // B*S*D threads
  long long m = t >> 10;            // row (b*S+s)
  int d = (int)(t & 1023);
  int s = (int)(m & (Ssz - 1));
  X[t] = emb[(long long)ids[m] * Dsz + d] + pos[(long long)s * Dsz + d];
}

__global__ __launch_bounds__(256) void ln_kernel(
    const float* __restrict__ in, const float* __restrict__ g,
    const float* __restrict__ bb, float* __restrict__ out)
{
  __shared__ float sh[16];
  long long row = blockIdx.x;
  const float* x = in + row * Dsz;
  float s = 0.f, s2 = 0.f;
  for (int d = threadIdx.x; d < Dsz; d += 256) { float v = x[d]; s += v; s2 += v * v; }
  s = waveSum(s); s2 = waveSum(s2);
  if ((threadIdx.x & 31) == 0) { sh[threadIdx.x >> 5] = s; sh[8 + (threadIdx.x >> 5)] = s2; }
  __syncthreads();
  float ts = 0.f, ts2 = 0.f;
#pragma unroll
  for (int i = 0; i < 8; ++i) { ts += sh[i]; ts2 += sh[8 + i]; }
  float mean = ts / Dsz;
  float var  = ts2 / Dsz - mean * mean;
  float inv  = rsqrtf(var + 1e-5f);
  float* o = out + row * Dsz;
  for (int d = threadIdx.x; d < Dsz; d += 256)
    o[d] = (x[d] - mean) * inv * g[d] + bb[d];
}

__global__ __launch_bounds__(256) void rotary_kernel(float* __restrict__ qkv)
{
  long long t = (long long)blockIdx.x * 256 + threadIdx.x;   // ROWS*H*64 threads
  if (t >= (long long)ROWS * Hsz * 64) return;
  int j = (int)(t & 63);
  int h = (int)((t >> 6) & (Hsz - 1));
  long long m = t >> 9;
  int s = (int)(m & (Ssz - 1));
  float invf = __expf(-(float)(2 * j) * (1.0f / 128.0f) * __logf(10000.0f));
  float ang = (float)s * invf;
  float sn, cs;
  __sincosf(ang, &sn, &cs);
  float* qp = qkv + m * 3072 + h * HDsz;
  float* kp = qp + Dsz;
  float q1 = qp[j], q2 = qp[j + 64];
  qp[j]      = q1 * cs - q2 * sn;
  qp[j + 64] = q2 * cs + q1 * sn;
  float k1 = kp[j], k2 = kp[j + 64];
  kp[j]      = k1 * cs - k2 * sn;
  kp[j + 64] = k2 * cs + k1 * sn;
}

__global__ __launch_bounds__(256) void attn_softmax_kernel(
    float* __restrict__ attn, const float* __restrict__ rel)
{
  __shared__ float sh[8];
  long long idx = blockIdx.x;                 // b*H*S + h*S + q
  int q = (int)(idx & (Ssz - 1));
  int h = (int)((idx >> 9) & (Hsz - 1));
  float* row = attn + idx * Ssz;
  int t = threadIdx.x;
  float v0 = row[t]       + rel[(q - t + Ssz - 1) * Hsz + h];
  float v1 = row[t + 256] + rel[(q - (t + 256) + Ssz - 1) * Hsz + h];
  float mx = waveMax(fmaxf(v0, v1));
  if ((t & 31) == 0) sh[t >> 5] = mx;
  __syncthreads();
  float m = sh[0];
#pragma unroll
  for (int i = 1; i < 8; ++i) m = fmaxf(m, sh[i]);
  __syncthreads();
  float e0 = __expf(v0 - m), e1 = __expf(v1 - m);
  float s = waveSum(e0 + e1);
  if ((t & 31) == 0) sh[t >> 5] = s;
  __syncthreads();
  float ts = 0.f;
#pragma unroll
  for (int i = 0; i < 8; ++i) ts += sh[i];
  float r = 1.0f / ts;
  row[t] = e0 * r;
  row[t + 256] = e1 * r;
}

__global__ __launch_bounds__(256) void router_kernel(
    const float* __restrict__ hn2, const float* __restrict__ Wr,
    const float* __restrict__ br, float* __restrict__ masks,
    float* __restrict__ rstat)
{
  int lane = threadIdx.x & 31;
  long long row = (long long)blockIdx.x * 8 + (threadIdx.x >> 5);
  if (row >= ROWS) return;
  float acc[Esz] = {};
  const float* x = hn2 + row * Dsz;
  for (int d = lane; d < Dsz; d += 32) {
    float h = x[d];
    const float* w = Wr + (long long)d * Esz;
#pragma unroll
    for (int e = 0; e < Esz; ++e) acc[e] += h * w[e];
  }
#pragma unroll
  for (int e = 0; e < Esz; ++e) acc[e] = waveSum(acc[e]);
  if (lane == 0) {
    float p[Esz], mx = -1e30f;
#pragma unroll
    for (int e = 0; e < Esz; ++e) { p[e] = acc[e] + br[e]; mx = fmaxf(mx, p[e]); }
    float sum = 0.f;
#pragma unroll
    for (int e = 0; e < Esz; ++e) { p[e] = __expf(p[e] - mx); sum += p[e]; }
    float inv = 1.0f / sum;
#pragma unroll
    for (int e = 0; e < Esz; ++e) p[e] *= inv;
    int i1 = 0;
#pragma unroll
    for (int e = 1; e < Esz; ++e) if (p[e] > p[i1]) i1 = e;
    int i2 = (i1 == 0) ? 1 : 0;
#pragma unroll
    for (int e = 0; e < Esz; ++e) if (e != i1 && p[e] > p[i2]) i2 = e;
    float* mrow = masks + row * Esz;
#pragma unroll
    for (int e = 0; e < Esz; ++e)
      mrow[e] = (e == i1) ? p[i1] : ((e == i2) ? p[i2] : 0.f);
    atomicAdd(&rstat[i1], 1.0f);
    atomicAdd(&rstat[i2], 1.0f);
#pragma unroll
    for (int e = 0; e < Esz; ++e) atomicAdd(&rstat[8 + e], p[e]);
  }
}

__global__ void zero16_kernel(float* p) { if (threadIdx.x < 16) p[threadIdx.x] = 0.f; }

__global__ void router_loss_kernel(const float* __restrict__ rstat, float* __restrict__ out)
{
  if (threadIdx.x == 0) {
    const float invR = 1.0f / (float)ROWS;
    float l = 0.f;
    for (int e = 0; e < Esz; ++e) l += (rstat[e] * invR) * (rstat[8 + e] * invR);
    out[0] = (float)Esz * l;
  }
}

__global__ __launch_bounds__(256) void feat_kernel(
    const float* __restrict__ eo, float* __restrict__ feat)
{
  int t = blockIdx.x * 256 + threadIdx.x;   // B*D threads
  int b = t >> 10, d = t & 1023;
  float s = 0.f;
  const float* base = eo + (long long)b * Ssz * Dsz + d;
  for (int i = 0; i < Ssz; ++i) s += base[(long long)i * Dsz];
  feat[t] = s * (1.0f / (float)Ssz);
}

__global__ __launch_bounds__(256) void cls_kernel(
    const float* __restrict__ feat, const float* __restrict__ Wc,
    const float* __restrict__ bc, float* __restrict__ cls)
{
  int t = blockIdx.x * 256 + threadIdx.x;   // B*D threads
  int b = t >> 10, n = t & 1023;
  float s = bc[n];
  const float* f = feat + (long long)b * Dsz;
  for (int d = 0; d < Dsz; ++d) s += f[d] * Wc[(long long)d * Dsz + n];
  cls[t] = s;
}

// ---------------------------------------------------------------------------
// Host launcher
// ---------------------------------------------------------------------------
extern "C" void kernel_launch(void* const* d_in, const int* in_sizes, int n_in,
                              void* d_out, int out_size, void* d_ws, size_t ws_size,
                              hipStream_t stream)
{
  (void)in_sizes; (void)n_in; (void)out_size; (void)ws_size;
  const int*   ids  = (const int*)  d_in[0];
  const float* emb  = (const float*)d_in[1];
  const float* pos  = (const float*)d_in[2];
  const float* ln1g = (const float*)d_in[3];
  const float* ln1b = (const float*)d_in[4];
  const float* ln2g = (const float*)d_in[5];
  const float* ln2b = (const float*)d_in[6];
  const float* ln3g = (const float*)d_in[7];
  const float* ln3b = (const float*)d_in[8];
  const float* Wqkv = (const float*)d_in[9];
  const float* bqkv = (const float*)d_in[10];
  const float* Wo   = (const float*)d_in[11];
  const float* bo   = (const float*)d_in[12];
  const float* rel  = (const float*)d_in[13];
  const float* Wr   = (const float*)d_in[14];
  const float* br   = (const float*)d_in[15];
  const float* We1  = (const float*)d_in[16];
  const float* be1  = (const float*)d_in[17];
  const float* We2  = (const float*)d_in[18];
  const float* be2  = (const float*)d_in[19];
  const float* Wc   = (const float*)d_in[20];
  const float* bc   = (const float*)d_in[21];

  float* out = (float*)d_out;
  float* ws  = (float*)d_ws;

  // Workspace layout (elements); buffers reused once their producers retire.
  float* X   = ws;                 // [8192,1024]; later MoE accumulator
  float* HN  = ws + 8388608LL;     // [8192,1024] hn / hn2
  float* QKV = ws + 16777216LL;    // [8192,3072]
  float* HB  = ws + 16777216LL;    // [8192,4096] MoE hidden (reuses QKV+SA)
  float* SA  = ws + 41943040LL;    // [8192,1024]
  float* X2  = ws + 50331648LL;    // [8192,1024]
  float* RST = ws + 58720256LL;    // 16 floats: f-count[7], p-sum[7]

  // Output layout (flat, return order)
  float* FEAT = out;                 // [16,1024]
  float* CLS  = out + 16384;         // [16,1024]
  float* LOSS = out + 32768;         // scalar
  float* ATTN = out + 32769;         // [16,8,512,512]
  float* EO   = out + 33587201LL;    // [16,512,1024]
  float* MASK = out + 41975809LL;    // [16,512,7]

  dim3 blk(256);

  // 1) embedding + positional
  embed_kernel<<<(ROWS * Dsz) / 256, blk, 0, stream>>>(ids, emb, pos, X);
  // 2) LN1
  ln_kernel<<<ROWS, blk, 0, stream>>>(X, ln1g, ln1b, HN);
  // 3) QKV projection [8192,1024]x[1024,3072]
  gemm_wmma_kernel<false, EPI_STORE, true><<<dim3(3072 / BN, ROWS / BM, 1), blk, 0, stream>>>(
      HN, Wqkv, bqkv, nullptr, nullptr, 0, QKV,
      Dsz, Dsz, 3072, 3072, 1.0f, 1, 0, 0, 0, 0, 0, 0);
  // 4) rotary on q,k (in place)
  rotary_kernel<<<(ROWS * Hsz * 64) / 256, blk, 0, stream>>>(QKV);
  // 5) scores = q k^T / sqrt(128), batched over 128 (b,h)
  gemm_wmma_kernel<true, EPI_STORE, true><<<dim3(Ssz / BN, Ssz / BM, Bsz * Hsz), blk, 0, stream>>>(
      QKV, QKV + Dsz, nullptr, nullptr, nullptr, 0, ATTN,
      HDsz, 3072, 3072, Ssz, 0.08838834764831845f, Hsz,
      1572864LL, 128LL, 1572864LL, 128LL, 2097152LL, 262144LL);
  // 6) softmax(scores + rel bias) -> attn (output)
  attn_softmax_kernel<<<Bsz * Hsz * Ssz, blk, 0, stream>>>(ATTN, rel);
  // 7) sa = attn @ v, batched (A from d_out slice: 4B-aligned -> scalar A stage)
  gemm_wmma_kernel<false, EPI_STORE, false><<<dim3(HDsz / BN, Ssz / BM, Bsz * Hsz), blk, 0, stream>>>(
      ATTN, QKV + 2 * Dsz, nullptr, nullptr, nullptr, 0, SA,
      Ssz, Ssz, 3072, Dsz, 1.0f, Hsz,
      2097152LL, 262144LL, 1572864LL, 128LL, 524288LL, 128LL);
  // 8) x2 = x + sa @ Wo + bo
  gemm_wmma_kernel<false, EPI_RESID, true><<<dim3(Dsz / BN, ROWS / BM, 1), blk, 0, stream>>>(
      SA, Wo, bo, X, nullptr, 0, X2,
      Dsz, Dsz, Dsz, Dsz, 1.0f, 1, 0, 0, 0, 0, 0, 0);
  // 9) LN2 + router (masks -> d_out, stats -> RST) + loss
  ln_kernel<<<ROWS, blk, 0, stream>>>(X2, ln2g, ln2b, HN);
  zero16_kernel<<<1, 32, 0, stream>>>(RST);
  router_kernel<<<ROWS / 8, blk, 0, stream>>>(HN, Wr, br, MASK, RST);
  router_loss_kernel<<<1, 32, 0, stream>>>(RST, LOSS);
  // 10) dense MoE: expert GEMM1 (+gelu) then mask-scaled GEMM2 accumulate
  for (int e = 0; e < Esz; ++e) {
    gemm_wmma_kernel<false, EPI_GELU, true><<<dim3(DFsz / BN, ROWS / BM, 1), blk, 0, stream>>>(
        X2, We1 + (long long)e * Dsz * DFsz, be1 + (long long)e * DFsz,
        nullptr, nullptr, 0, HB,
        Dsz, Dsz, DFsz, DFsz, 1.0f, 1, 0, 0, 0, 0, 0, 0);
    if (e == 0) {
      gemm_wmma_kernel<false, EPI_MASK_INIT, true><<<dim3(Dsz / BN, ROWS / BM, 1), blk, 0, stream>>>(
          HB, We2 + (long long)e * DFsz * Dsz, be2 + (long long)e * Dsz,
          nullptr, MASK + e, Esz, X /*ACC*/,
          DFsz, DFsz, Dsz, Dsz, 1.0f, 1, 0, 0, 0, 0, 0, 0);
    } else {
      gemm_wmma_kernel<false, EPI_MASK_ADD, true><<<dim3(Dsz / BN, ROWS / BM, 1), blk, 0, stream>>>(
          HB, We2 + (long long)e * DFsz * Dsz, be2 + (long long)e * Dsz,
          nullptr, MASK + e, Esz, X /*ACC*/,
          DFsz, DFsz, Dsz, Dsz, 1.0f, 1, 0, 0, 0, 0, 0, 0);
    }
  }
  // 11) eo = LN3(acc) -> d_out; feat = mean_S(eo); cls = feat @ Wc + bc
  ln_kernel<<<ROWS, blk, 0, stream>>>(X, ln3g, ln3b, EO);
  feat_kernel<<<(Bsz * Dsz) / 256, blk, 0, stream>>>(EO, FEAT);
  cls_kernel<<<(Bsz * Dsz) / 256, blk, 0, stream>>>(FEAT, Wc, bc, CLS);
}
